// ExportGatedDeltaNet_63324997812733
// MI455X (gfx1250) — matrix operand order repaired and checked by reading, not compiled, not run on hardware
//
#include <hip/hip_runtime.h>
#include <hip/hip_bf16.h>
#include <math.h>

// ---------------- problem constants ----------------
#define Bb 2
#define Tt 2048
#define Cc 2048
#define Hh 16
#define DKk 128
#define DVv 128
#define KEY_DIM 2048
#define VALUE_DIM 2048
#define CONV_DIM 6144
#define NQKVZ 8192
#define Mrows 4096

typedef __attribute__((ext_vector_type(16))) __bf16 v16bf;
typedef __attribute__((ext_vector_type(8)))  __bf16 v8bf;
typedef __attribute__((ext_vector_type(8)))  float  v8f;

__device__ __forceinline__ float sigmoidf_(float x) { return 1.0f / (1.0f + __expf(-x)); }

__device__ __forceinline__ v8f wmma_bf16(v16bf a, v16bf b, v8f c) {
  return __builtin_amdgcn_wmma_f32_16x16x32_bf16(false, a, false, b, (short)0, c, false, false);
}

// ---- gfx1250 async global->LDS copy (ASYNCcnt-tracked), 16 bytes per lane ----
__device__ __forceinline__ void async_copy16(const void* lds_dst, const void* gsrc) {
  unsigned lds_off = (unsigned)(size_t)lds_dst;                 // low 32 bits of the pointer give the LDS address
  unsigned long long ga = (unsigned long long)(size_t)gsrc;
  asm volatile("global_load_async_to_lds_b128 %0, %1, off" :: "v"(lds_off), "v"(ga) : "memory");
}
__device__ __forceinline__ void wait_async0() {
  asm volatile("s_wait_asynccnt 0" ::: "memory");
}

// ---------------- LDS fragment gathers (wave32, contiguous k = 16*half + i per lane half) ----------------
// A frag (16x32): LDS [row][k]; contiguous 16-half run per lane half
__device__ __forceinline__ v16bf ldsA(const __bf16* T, int m0, int k0, int ld, int lane) {
  int r  = m0 + (lane & 15);
  int kh = k0 + ((lane & 16) ? 16 : 0);
  const __bf16* p = T + r * ld + kh;
  v16bf f;
#pragma unroll
  for (int i = 0; i < 16; ++i) f[i] = p[i];
  return f;
}
// B frag (32x16): logical B[k][n] stored as [n][k] -> contiguous 16-half run
__device__ __forceinline__ v16bf ldsB_nk(const __bf16* T, int n0, int k0, int ld, int lane) {
  int n  = n0 + (lane & 15);
  int kh = k0 + ((lane & 16) ? 16 : 0);
  const __bf16* p = T + n * ld + kh;
  v16bf f;
#pragma unroll
  for (int i = 0; i < 16; ++i) f[i] = p[i];
  return f;
}
// B frag from f32 row-major [k][n] (state matrix), converted on gather
__device__ __forceinline__ v16bf ldsB_kn_f32(const float* T, int n0, int k0, int ld, int lane) {
  int n  = n0 + (lane & 15);
  int kh = k0 + ((lane & 16) ? 16 : 0);
  v16bf f;
#pragma unroll
  for (int i = 0; i < 16; ++i) f[i] = (__bf16)T[(kh + i) * ld + n];
  return f;
}

// ================= f32 -> bf16 conversion pre-pass =================
__global__ void cvt_bf16(const float* __restrict__ src, __bf16* __restrict__ dst, size_t n) {
  size_t i = ((size_t)blockIdx.x * 256 + threadIdx.x) * 4;
  if (i < n) {
    if (i + 4096 < n) __builtin_prefetch(src + i + 4096, 0, 0);
    float4 v = *(const float4*)(src + i);
    dst[i + 0] = (__bf16)v.x; dst[i + 1] = (__bf16)v.y;
    dst[i + 2] = (__bf16)v.z; dst[i + 3] = (__bf16)v.w;
  }
}

// ===== WMMA bf16 GEMM, double-buffered (A async-staged, B transposed-staged) =====
#define BM 128
#define BN 128
#define BK 32
#define LDT2 40    // tile stride (halfs): 80 B row (16B-multiple for async A)

__global__ __launch_bounds__(256)
void gemm_bf16_wmma(const __bf16* __restrict__ A, const __bf16* __restrict__ Bm,
                    float* __restrict__ C, int M, int N, int K) {
  (void)M;
  __shared__ __bf16 lds_a[2][BM * LDT2];   // [m][k]
  __shared__ __bf16 lds_b[2][BN * LDT2];   // [n][k]

  const int tid  = threadIdx.x;
  const int lane = tid & 31;
  const int wave = tid >> 5;
  const int n0 = blockIdx.x * BN;
  const int m0 = blockIdx.y * BM;
  const int wm = (wave & 3) * 32;
  const int wn = (wave >> 2) * 64;

  v8f acc[2][4];
#pragma unroll
  for (int i = 0; i < 2; ++i)
#pragma unroll
    for (int j = 0; j < 4; ++j) acc[i][j] = (v8f)(0.0f);

  auto stage = [&](int p, int k0) {
    // A 128x32: async 16B chunks, 4 per row
#pragma unroll
    for (int i = 0; i < 2; ++i) {
      int idx = tid + i * 256;                   // 0..511
      int r = idx >> 2, c = idx & 3;
      async_copy16(&lds_a[p][r * LDT2 + c * 8],
                   &A[(size_t)(m0 + r) * K + k0 + c * 8]);
    }
    // B 32x128: b128 global loads along n, transpose into [n][k] with b16 stores
#pragma unroll
    for (int i = 0; i < 2; ++i) {
      int idx = tid + i * 256;                   // 0..511
      int kr = idx >> 4, c8 = idx & 15;
      v8bf vv = *(const v8bf*)(&Bm[(size_t)(k0 + kr) * N + n0 + c8 * 8]);
#pragma unroll
      for (int e = 0; e < 8; ++e)
        lds_b[p][(c8 * 8 + e) * LDT2 + kr] = vv[e];
    }
  };

  stage(0, 0);
  int p = 0;
  for (int k0 = 0; k0 < K; k0 += BK) {
    wait_async0();
    __syncthreads();
    if (k0 + BK < K) stage(p ^ 1, k0 + BK);

    v16bf af[2], bf_[4];
#pragma unroll
    for (int i = 0; i < 2; ++i) af[i] = ldsA(lds_a[p], wm + 16 * i, 0, LDT2, lane);
#pragma unroll
    for (int j = 0; j < 4; ++j) bf_[j] = ldsB_nk(lds_b[p], wn + 16 * j, 0, LDT2, lane);
#pragma unroll
    for (int i = 0; i < 2; ++i)
#pragma unroll
      for (int j = 0; j < 4; ++j)
        acc[i][j] = wmma_bf16(af[i], bf_[j], acc[i][j]);
    p ^= 1;
  }

#pragma unroll
  for (int i = 0; i < 2; ++i)
#pragma unroll
    for (int j = 0; j < 4; ++j) {
      int rowb = m0 + wm + 16 * i + ((lane >> 4) << 3);
      int col  = n0 + wn + 16 * j + (lane & 15);
#pragma unroll
      for (int r = 0; r < 8; ++r)
        C[(size_t)(rowb + r) * N + col] = acc[i][j][r];
    }
}

// ================= alpha / beta =================
__global__ void ab_kernel(const float* __restrict__ x, const float* __restrict__ Wb,
                          const float* __restrict__ Wa, float* __restrict__ ab) {
  int row  = blockIdx.x;
  int lane = threadIdx.x;
  const float* W = (lane < 16) ? Wb : Wa;
  int col = lane & 15;
  float acc = 0.0f;
  const float* xr = x + (size_t)row * Cc;
  for (int k = 0; k < Cc; ++k) acc += xr[k] * W[k * Hh + col];
  ab[(size_t)row * 32 + lane] = sigmoidf_(acc);
}

// ============ depthwise causal conv (K=4) + SiLU + per-head L2 norm -> bf16 ============
__global__ __launch_bounds__(256)
void conv_silu_norm(const float* __restrict__ qkvz, const float* __restrict__ conv_w,
                    const float* __restrict__ conv_state, const int* __restrict__ input_pos,
                    __bf16* __restrict__ y) {
  __shared__ float red[256];
  int row = blockIdx.x;
  int b = row / Tt, t = row % Tt;
  int c = blockIdx.y * 256 + threadIdx.x;
  int tid = threadIdx.x;
  float keep = (input_pos[0] == 0) ? 0.0f : 1.0f;

  float w[4];
#pragma unroll
  for (int j = 0; j < 4; ++j) w[j] = conv_w[c * 4 + j];

  float acc = 0.0f;
#pragma unroll
  for (int j = 0; j < 4; ++j) {
    int ts = t + j - 3;
    float inv;
    if (ts >= 0) inv = qkvz[(size_t)(b * Tt + ts) * NQKVZ + c];
    else         inv = keep * conv_state[((size_t)b * CONV_DIM + c) * 4 + (t + j + 1)];
    acc += w[j] * inv;
  }
  float sv = acc * sigmoidf_(acc);
  float out = sv;

  if (blockIdx.y < 16) {
    red[tid] = sv * sv;
    __syncthreads();
#pragma unroll
    for (int s = 64; s > 0; s >>= 1) {
      if ((tid & 127) < s) red[tid] += red[tid + s];
      __syncthreads();
    }
    float nrm = fmaxf(sqrtf(red[tid & ~127]), 1e-12f);
    out = sv / nrm;
  }
  y[(size_t)row * CONV_DIM + c] = (__bf16)out;
}

// ====== chunked gated scan on WMMA (async-staged); fused RMS-norm + sigmoid(z) gate ======
#define CL 64
#define LDQ 136    // 272 B rows: 16B-multiple for async B128 destinations
#define LDATT 66
#define LDT 66     // transposed k/v tiles: [d][t]

__global__ __launch_bounds__(256)
void scan_chunked(const __bf16* __restrict__ y, const float* __restrict__ ab,
                  const float* __restrict__ qkvz, const float* __restrict__ rec,
                  const int* __restrict__ input_pos, const float* __restrict__ norm_w,
                  __bf16* __restrict__ gated) {
  __shared__ float  Sf[DKk * DVv];          // 64 KB persistent state (f32)
  __shared__ __bf16 qraw[CL * LDQ];         // [t][dk]
  __shared__ __bf16 kraw[CL * LDQ];         // [t][dk]
  __shared__ __bf16 kT[DKk * LDT];          // k-hat transposed: [dk][t]
  __shared__ __bf16 vT[DVv * LDT];          // v transposed:     [dv][t]
  __shared__ __bf16 attb[CL * LDATT];       // [t][j]
  __shared__ __bf16 outfb[CL * DVv];
  __shared__ float  la[CL], lb[CL], rs[CL];

  const int bh = blockIdx.x;
  const int b = bh >> 4, h = bh & 15;
  const int tid = threadIdx.x, lane = tid & 31, w = tid >> 5;
  const float keep = (input_pos[0] == 0) ? 0.0f : 1.0f;

  for (int i = tid; i < DKk * DVv; i += 256)
    Sf[i] = keep * rec[(size_t)bh * DKk * DVv + i];
  __syncthreads();

  for (int c0 = 0; c0 < Tt; c0 += CL) {
    // ---- async-stage q,k tiles ----
#pragma unroll
    for (int i = 0; i < 8; ++i) {
      int idx = tid + i * 256;                 // 0..2047
      int mm  = idx >> 10;                     // 0=q,1=k
      int rem = idx & 1023;
      int t = rem >> 4, c8 = rem & 15;
      __bf16* dst = (mm ? kraw : qraw) + t * LDQ + c8 * 8;
      const __bf16* g = y + (size_t)(b * Tt + c0 + t) * CONV_DIM + mm * KEY_DIM + h * 128 + c8 * 8;
      async_copy16(dst, g);
    }
    // ---- stage v transposed ([dv][t]) via b128 loads + b16 scatter ----
#pragma unroll
    for (int i = 0; i < 4; ++i) {
      int idx = tid + i * 256;                 // 0..1023
      int t = idx >> 4, c8 = idx & 15;
      v8bf vv = *(const v8bf*)(y + (size_t)(b * Tt + c0 + t) * CONV_DIM + 2 * KEY_DIM + h * 128 + c8 * 8);
#pragma unroll
      for (int e = 0; e < 8; ++e) vT[(c8 * 8 + e) * LDT + t] = vv[e];
    }

    // ---- decay coefficients: la = cumsum(log alpha), lb = beta ----
    if (tid < CL) {
      size_t row = (size_t)(b * Tt + c0 + tid) * 32;
      la[tid] = __logf(ab[row + 16 + h]);
      lb[tid] = ab[row + h];
    }
    __syncthreads();
#pragma unroll
    for (int off = 1; off < CL; off <<= 1) {
      float tmp = 0.0f;
      if (tid < CL && tid >= off) tmp = la[tid - off];
      __syncthreads();
      if (tid < CL && tid >= off) la[tid] += tmp;
      __syncthreads();
    }
    wait_async0();
    __syncthreads();
    const float la63 = la[CL - 1];

    // ---- GEMM2: att_raw = Q K^T (64x64, K=128); mask in regs; store bf16 ----
    {
      const int mt = (w & 3) * 16, nt = (w >> 2) * 32;
      v8f a2[2]; a2[0] = (v8f)(0.0f); a2[1] = (v8f)(0.0f);
#pragma unroll
      for (int k0 = 0; k0 < 128; k0 += 32) {
        v16bf af = ldsA(qraw, mt, k0, LDQ, lane);
#pragma unroll
        for (int j = 0; j < 2; ++j)
          a2[j] = wmma_bf16(af, ldsB_nk(kraw, nt + 16 * j, k0, LDQ, lane), a2[j]);
      }
      const int tb = mt + ((lane >> 4) << 3);
#pragma unroll
      for (int j = 0; j < 2; ++j) {
        int jj = nt + 16 * j + (lane & 15);
#pragma unroll
        for (int r = 0; r < 8; ++r) {
          int t = tb + r;
          float wgt = (jj <= t) ? lb[jj] * __expf(la[t] - la[jj]) : 0.0f;
          attb[t * LDATT + jj] = (__bf16)(a2[j][r] * wgt);
        }
      }
    }
    // ---- K-hat transposed: kT[dk][t] = k[t][dk] * beta_t * exp(la63 - la_t) ----
    for (int i = 0; i < (CL * 128) / 256; ++i) {
      int idx = tid + i * 256;
      int t = idx >> 7, dk = idx & 127;
      float cf = lb[t] * __expf(la63 - la[t]);
      kT[dk * LDT + t] = (__bf16)((float)kraw[t * LDQ + dk] * cf);
    }
    __syncthreads();

    // ---- out = diag(exp(la)) * (Q @ S_prev) + Att @ V  (64x128) ----
    {
      const int mt = (w & 1) * 32, nt = (w >> 1) * 32;
      v8f po[2][2];
#pragma unroll
      for (int i = 0; i < 2; ++i)
#pragma unroll
        for (int j = 0; j < 2; ++j) po[i][j] = (v8f)(0.0f);

#pragma unroll
      for (int k0 = 0; k0 < 128; k0 += 32) {
        v16bf af[2];
#pragma unroll
        for (int i = 0; i < 2; ++i) af[i] = ldsA(qraw, mt + 16 * i, k0, LDQ, lane);
#pragma unroll
        for (int j = 0; j < 2; ++j) {
          v16bf bf_ = ldsB_kn_f32(Sf, nt + 16 * j, k0, DVv, lane);
#pragma unroll
          for (int i = 0; i < 2; ++i) po[i][j] = wmma_bf16(af[i], bf_, po[i][j]);
        }
      }
#pragma unroll
      for (int i = 0; i < 2; ++i) {
        int tb = mt + 16 * i + ((lane >> 4) << 3);
#pragma unroll
        for (int r = 0; r < 8; ++r) {
          float e = __expf(la[tb + r]);
#pragma unroll
          for (int j = 0; j < 2; ++j) po[i][j][r] *= e;
        }
      }
#pragma unroll
      for (int k0 = 0; k0 < CL; k0 += 32) {
        v16bf af[2];
#pragma unroll
        for (int i = 0; i < 2; ++i) af[i] = ldsA(attb, mt + 16 * i, k0, LDATT, lane);
#pragma unroll
        for (int j = 0; j < 2; ++j) {
          v16bf bf_ = ldsB_nk(vT, nt + 16 * j, k0, LDT, lane);
#pragma unroll
          for (int i = 0; i < 2; ++i) po[i][j] = wmma_bf16(af[i], bf_, po[i][j]);
        }
      }
#pragma unroll
      for (int i = 0; i < 2; ++i)
#pragma unroll
        for (int j = 0; j < 2; ++j) {
          int rowb = mt + 16 * i + ((lane >> 4) << 3);
          int col  = nt + 16 * j + (lane & 15);
#pragma unroll
          for (int r = 0; r < 8; ++r)
            outfb[(rowb + r) * DVv + col] = (__bf16)po[i][j][r];
        }
    }

    // ---- GEMM4: S = exp(la63)*S + K-hat^T @ V  (128x128, K=64) ----
    {
      const int mt = (w & 3) * 32, nt = (w >> 2) * 64;
      v8f as[2][4];
#pragma unroll
      for (int i = 0; i < 2; ++i)
#pragma unroll
        for (int j = 0; j < 4; ++j) as[i][j] = (v8f)(0.0f);
#pragma unroll
      for (int k0 = 0; k0 < CL; k0 += 32) {
        v16bf af[2];
#pragma unroll
        for (int i = 0; i < 2; ++i) af[i] = ldsA(kT, mt + 16 * i, k0, LDT, lane);
#pragma unroll
        for (int j = 0; j < 4; ++j) {
          v16bf bf_ = ldsB_nk(vT, nt + 16 * j, k0, LDT, lane);
#pragma unroll
          for (int i = 0; i < 2; ++i) as[i][j] = wmma_bf16(af[i], bf_, as[i][j]);
        }
      }
      __syncthreads();  // all S_prev reads complete before overwriting Sf
      const float decay = __expf(la63);
#pragma unroll
      for (int i = 0; i < 2; ++i)
#pragma unroll
        for (int j = 0; j < 4; ++j) {
          int rowb = mt + 16 * i + ((lane >> 4) << 3);
          int col  = nt + 16 * j + (lane & 15);
#pragma unroll
          for (int r = 0; r < 8; ++r) {
            int idx = (rowb + r) * DVv + col;
            Sf[idx] = decay * Sf[idx] + as[i][j][r];
          }
        }
    }
    __syncthreads();

    // ---- epilogue: RMS-norm + sigmoid(z) gate -> bf16 ----
    if (tid < CL) {
      float s = 0.0f;
      for (int dv = 0; dv < DVv; ++dv) { float v = (float)outfb[tid * DVv + dv]; s += v * v; }
      rs[tid] = rsqrtf(s * (1.0f / DVv) + 1e-6f);
    }
    __syncthreads();
    for (int i = 0; i < (CL * 128) / 256; ++i) {
      int idx = tid + i * 256;
      int t = idx >> 7, dv = idx & 127;
      size_t row = (size_t)(b * Tt + c0 + t);
      float z = qkvz[row * NQKVZ + CONV_DIM + h * 128 + dv];
      gated[row * VALUE_DIM + h * 128 + dv] =
          (__bf16)((float)outfb[idx] * rs[t] * norm_w[dv] * sigmoidf_(z));
    }
    __syncthreads();
  }
}

// =================================== launcher ===================================
extern "C" void kernel_launch(void* const* d_in, const int* in_sizes, int n_in,
                              void* d_out, int out_size, void* d_ws, size_t ws_size,
                              hipStream_t stream) {
  (void)in_sizes; (void)n_in; (void)out_size; (void)ws_size;
  const float* x          = (const float*)d_in[0];
  const int*   input_pos  = (const int*)  d_in[1];
  const float* W_qkvz     = (const float*)d_in[2];
  const float* W_b        = (const float*)d_in[3];
  const float* W_a        = (const float*)d_in[4];
  const float* conv_w     = (const float*)d_in[5];
  const float* norm_w     = (const float*)d_in[6];
  const float* W_out      = (const float*)d_in[7];
  const float* conv_state = (const float*)d_in[8];
  const float* rec_state  = (const float*)d_in[9];
  float* out = (float*)d_out;

  char* ws = (char*)d_ws;
  float*  qkvz   = (float*) (ws);                 // 128 MB f32
  __bf16* yconv  = (__bf16*)(ws + 134217728ull);  //  48 MB bf16
  float*  ab     = (float*) (ws + 184549376ull);  // 0.5 MB
  __bf16* gated  = (__bf16*)(ws + 185073664ull);  //  16 MB bf16
  __bf16* xb     = (__bf16*)(ws + 201850880ull);  //  16 MB
  __bf16* Wqkvzb = (__bf16*)(ws + 218628096ull);  //  32 MB
  __bf16* Woutb  = (__bf16*)(ws + 252182528ull);  //   8 MB

  cvt_bf16<<<(Mrows * Cc) / 1024, 256, 0, stream>>>(x, xb, (size_t)Mrows * Cc);
  cvt_bf16<<<(Cc * NQKVZ) / 1024, 256, 0, stream>>>(W_qkvz, Wqkvzb, (size_t)Cc * NQKVZ);
  cvt_bf16<<<(VALUE_DIM * Cc) / 1024, 256, 0, stream>>>(W_out, Woutb, (size_t)VALUE_DIM * Cc);

  gemm_bf16_wmma<<<dim3(NQKVZ / BN, Mrows / BM), 256, 0, stream>>>(
      xb, Wqkvzb, qkvz, Mrows, NQKVZ, Cc);
  ab_kernel<<<Mrows, 32, 0, stream>>>(x, W_b, W_a, ab);
  conv_silu_norm<<<dim3(Mrows, CONV_DIM / 256), 256, 0, stream>>>(
      qkvz, conv_w, conv_state, input_pos, yconv);
  scan_chunked<<<Bb * Hh, 256, 0, stream>>>(
      yconv, ab, qkvz, rec_state, input_pos, norm_w, gated);
  gemm_bf16_wmma<<<dim3(Cc / BN, Mrows / BM), 256, 0, stream>>>(
      gated, Woutb, out, Mrows, Cc, Cc);
}
